// CompanionSSM_15547781611900
// MI455X (gfx1250) — matrix operand order repaired and checked by reading, not compiled, not run on hardware
//
#include <hip/hip_runtime.h>

// ---------------------------------------------------------------------------
// CompanionSSM on MI455X (gfx1250): causal conv as block-Toeplitz bf16 WMMA,
// with TDM (tensor_load_to_lds) staging of the Toeplitz kernel strip.
//   u: (4, 4096, 1024) f32   a,b,c: (256, 64) f32   out: (4, 4096, 1024) f32
// ---------------------------------------------------------------------------

typedef __bf16 bf16;
typedef __bf16 v16bf __attribute__((ext_vector_type(16)));
typedef float  v8f   __attribute__((ext_vector_type(8)));
typedef unsigned int u32x4 __attribute__((ext_vector_type(4)));
typedef int          i32x4 __attribute__((ext_vector_type(4)));
typedef int          i32x8 __attribute__((ext_vector_type(8)));
typedef unsigned int u4    __attribute__((ext_vector_type(4)));

#define NKc 256
#define KDc 64
#define Lc  4096
#define Bc  4
#define Dc  1024
#define COLSTRIDE 4112          // (4096 + 16 pad rows); x2B = 8224 B = 32B-aligned cols
#define KPADLEN 4128            // 15 leading zeros + 4096 + 17 tail zeros (16B multiple)

// ---------------------------------------------------------------------------
// Kernel 1: per-row L1 norms of a, global-mean threshold, per-row scale.
// ---------------------------------------------------------------------------
__global__ __launch_bounds__(256) void norm_kernel(const float* __restrict__ a,
                                                   float* __restrict__ f) {
  __shared__ float red[256];
  const int n = threadIdx.x;
  float s = 0.f;
  #pragma unroll 8
  for (int i = 0; i < KDc; ++i) s += fabsf(a[n * KDc + i]);
  red[n] = s;
  __syncthreads();
  for (int off = 128; off > 0; off >>= 1) {
    if (n < off) red[n] += red[n + off];
    __syncthreads();
  }
  const float mean = red[0] * (1.0f / 256.0f);
  f[n] = (mean > 1e-4f) ? (1.0f / s) : 1.0f;
}

// ---------------------------------------------------------------------------
// Kernel 2: k[n, 64q+r] = c^T (A^64)^q (A^r b), written PRE-PADDED:
//   kbf[n][0..14]=0 | kbf[n][15+l]=k[l] | tail zeros  (KPADLEN bf16 per n)
// so the conv kernel can DMA it into LDS as a raw strip via the TDM.
// ---------------------------------------------------------------------------
__global__ __launch_bounds__(64) void build_k(const float* __restrict__ a,
                                              const float* __restrict__ b,
                                              const float* __restrict__ c,
                                              const float* __restrict__ f,
                                              bf16* __restrict__ kbf) {
  const int n = blockIdx.x;
  const int i = threadIdx.x;           // state element 0..63
  __shared__ float Pl[KDc * (KDc + 1)];
  __shared__ float Ul[KDc * (KDc + 1)];
  __shared__ float s[KDc];
  __shared__ float w[KDc];

  const float an = a[n * KDc + i] * f[n];
  const float bi = b[n * KDc + i];
  const float ci = c[n * KDc + i];

  bf16* kn = kbf + (size_t)n * KPADLEN;
  if (i < 15) kn[i] = (bf16)0.f;                       // leading zero pad
  if (i < KPADLEN - 15 - Lc) kn[15 + Lc + i] = (bf16)0.f;  // tail pad

  // ---- U = [b, Ab, ..., A^63 b] (shift recurrence: s'[i]=s[i-1]+an[i]*s[63])
  s[i] = bi;
  __syncthreads();
  Ul[i * (KDc + 1) + 0] = bi;
  for (int r = 1; r < KDc; ++r) {
    const float s63 = s[KDc - 1];
    const float prev = (i > 0) ? s[i - 1] : 0.f;
    __syncthreads();
    const float ns = prev + an * s63;
    s[i] = ns;
    __syncthreads();
    Ul[i * (KDc + 1) + r] = ns;
  }

  // ---- P = A^64: states 64..127 of the shift recurrence from e0.
  s[i] = (i == 0) ? 1.f : 0.f;
  __syncthreads();
  for (int step = 1; step <= 127; ++step) {
    const float s63 = s[KDc - 1];
    const float prev = (i > 0) ? s[i - 1] : 0.f;
    __syncthreads();
    const float ns = prev + an * s63;
    s[i] = ns;
    __syncthreads();
    if (step >= 64) Pl[i * (KDc + 1) + (step - 64)] = ns;
  }

  // ---- k[64q + r] = w_q . U[:,r],  w_{q+1} = P^T w_q,  w_0 = c.
  w[i] = ci;
  __syncthreads();
  for (int q = 0; q < KDc; ++q) {
    float acc = 0.f;
    #pragma unroll 8
    for (int t = 0; t < KDc; ++t) acc += w[t] * Ul[t * (KDc + 1) + i];
    kn[15 + q * KDc + i] = (bf16)acc;
    float nw = 0.f;
    #pragma unroll 8
    for (int t = 0; t < KDc; ++t) nw += w[t] * Pl[t * (KDc + 1) + i];
    __syncthreads();
    w[i] = nw;
    __syncthreads();
  }
}

// ---------------------------------------------------------------------------
// Kernel 3: Y_n (4096x16) = LowerToeplitz(k_n) @ U_n via
// v_wmma_f32_16x16x32_bf16.  Kernel strip staged by the Tensor Data Mover
// (overlapped with the f32->bf16 u staging); activation strip LDS-resident.
// ---------------------------------------------------------------------------
__global__ __launch_bounds__(256) void conv_wmma(const float* __restrict__ u,
                                                 const bf16* __restrict__ kbf,
                                                 float* __restrict__ y) {
  extern __shared__ uint4 smem4[];
  bf16* u_lds = (bf16*)smem4;              // [16 cols][COLSTRIDE rows] col-major
  bf16* klds  = u_lds + 16 * COLSTRIDE;    // KPADLEN bf16 (15 zeros | k | zeros)

  const int n   = blockIdx.x >> 1;
  const int p   = blockIdx.x & 1;
  const int tid = threadIdx.x;

  // ---- kernel strip -> LDS via TDM (wave 0 issues; EXEC-independent DMA)
#if __has_builtin(__builtin_amdgcn_tensor_load_to_lds)
  const bool use_tdm = true;
  if (tid < 32) {
    const unsigned long long ga =
        (unsigned long long)(uintptr_t)(kbf + (size_t)n * KPADLEN);
    const unsigned int lds_off =
        __builtin_amdgcn_groupstaticsize() + 16u * COLSTRIDE * 2u;
    u32x4 g0;
    g0.x = 1u;                                    // count=1, user descriptor
    g0.y = lds_off;                               // lds_addr (bytes)
    g0.z = (unsigned int)(ga & 0xFFFFFFFFu);      // global_addr[31:0]
    g0.w = ((unsigned int)(ga >> 32) & 0x01FFFFFFu) | 0x80000000u; // addr hi | type=2
    i32x8 g1;
    g1[0] = 0x00010000;                 // workgroup_mask=0, data_size=1 (2B)
    g1[1] = (int)(KPADLEN << 16);       // tensor_dim0[15:0] in bits 63:48
    g1[2] = 0;                          // tensor_dim0 hi, tensor_dim1 lo
    g1[3] = (int)(KPADLEN << 16);       // tile_dim0 in bits 127:112
    g1[4] = 0;                          // tile_dim1=0, tile_dim2=0 (1D strip)
    g1[5] = KPADLEN;                    // tensor_dim0_stride[31:0]
    g1[6] = 0;
    g1[7] = 0;
    const i32x4 gz = {0, 0, 0, 0};
#if defined(__clang_major__) && (__clang_major__ >= 23)
    const i32x8 gz8 = {0, 0, 0, 0, 0, 0, 0, 0};
    __builtin_amdgcn_tensor_load_to_lds(g0, g1, gz, gz, gz8, 0);
#else
    __builtin_amdgcn_tensor_load_to_lds(g0, g1, gz, gz, 0);
#endif
  }
#else
  const bool use_tdm = false;
  for (int j = tid; j < KPADLEN; j += 256)
    klds[j] = kbf[(size_t)n * KPADLEN + j];
#endif

  // ---- zero the 16 causal-pad rows of each column (256 threads = 16x16)
  {
    const int col = tid >> 4, row = tid & 15;
    u_lds[col * COLSTRIDE + row] = (bf16)0.f;
  }
  // ---- stage u strip: cols = b*4 + h, d = 4n + h; float4 -> 4 bf16
  for (int it = 0; it < 64; ++it) {
    const int item = tid + 256 * it;            // 0..16383 = (b, t)
    const int bb = item >> 12;
    const int t  = item & (Lc - 1);
    const float* up = u + ((size_t)(bb * Lc + t)) * Dc + n * 4;
    __builtin_prefetch(up + 4 * Dc, 0, 0);      // global_prefetch_b8
    const float4 v = *(const float4*)up;
    const int cb = bb * 4;
    u_lds[(cb + 0) * COLSTRIDE + 16 + t] = (bf16)v.x;
    u_lds[(cb + 1) * COLSTRIDE + 16 + t] = (bf16)v.y;
    u_lds[(cb + 2) * COLSTRIDE + 16 + t] = (bf16)v.z;
    u_lds[(cb + 3) * COLSTRIDE + 16 + t] = (bf16)v.w;
  }
#if __has_builtin(__builtin_amdgcn_tensor_load_to_lds)
  if (tid < 32) __builtin_amdgcn_s_wait_tensorcnt(0);
#endif
  (void)use_tdm;
  __syncthreads();

  const int wv   = tid >> 5;                    // wave 0..7
  const int lane = tid & 31;
  const int vres = p * 8 + wv;                  // residue class 0..15
  const int r    = lane & 15;                   // A-matrix row
  const int kb8  = (lane & 16) ? 8 : 0;         // half the A lane-half K offset (2*kb8 = 16)
  const int col  = lane & 15;                   // B/C/D column
  // B layout: lanes 0-15 hold K=0..15 (u block i-2t), lanes 16-31 K=16..31
  // (u block i-2t-1); both collapse to one contiguous 16-bf16 run (32B-aligned).
  const bf16* ucol = u_lds + col * COLSTRIDE + ((lane & 16) ? 0 : 16);

  for (int chunk = 0; chunk < 2; ++chunk) {
    const int ibase = vres + 128 * chunk;       // tiles i = ibase + 16*s
    const v8f vzero = {0.f, 0.f, 0.f, 0.f, 0.f, 0.f, 0.f, 0.f};
    v8f acc[8];
    #pragma unroll
    for (int s2 = 0; s2 < 8; ++s2) acc[s2] = vzero;

    const int tmax = (ibase + 112) >> 1;        // largest tile's K-pair count
    for (int t = 0; t <= tmax; ++t) {
      // ---- A fragment: [T_{2t} | T_{2t+1}], T_m[r][c] = k[16m + r - c]
      v16bf afrag;
      const int base = 32 * t + r + 15;         // +15 = zero-pad offset
      #pragma unroll
      for (int e = 0; e < 16; ++e) {
        const int idx = base + 2 * kb8 - e;
        afrag[e] = klds[idx];
      }
      // ---- 8 output tiles reuse this A fragment (wave-uniform guards)
      #pragma unroll
      for (int s2 = 7; s2 >= 0; --s2) {
        const int i = ibase + 16 * s2;
        if (t > (i >> 1)) break;                // smaller s2 => smaller i
        const int blk = i - 2 * t;
        const u4* bp = (const u4*)(ucol + blk * 16);  // 2x ds_load_b128
        union { u4 u2[2]; v16bf v; } cvt;
        cvt.u2[0] = bp[0];
        cvt.u2[1] = bp[1];
        acc[s2] = __builtin_amdgcn_wmma_f32_16x16x32_bf16(
            false, afrag, false, cvt.v, (short)0, acc[s2], false, false);
      }
    }

    // ---- store D tiles: lane col -> (batch, head); element e -> time row
    const int bb = col >> 2, hh = col & 3;
    const int rowadd = (lane & 16) ? 8 : 0;
    const size_t didx = (size_t)(n * 4 + hh);
    #pragma unroll
    for (int s2 = 0; s2 < 8; ++s2) {
      const int i = ibase + 16 * s2;
      #pragma unroll
      for (int e = 0; e < 8; ++e) {
        const int q = i * 16 + e + rowadd;
        y[(size_t)bb * (Lc * Dc) + (size_t)q * Dc + didx] = acc[s2][e];
      }
    }
  }
}

// ---------------------------------------------------------------------------
extern "C" void kernel_launch(void* const* d_in, const int* in_sizes, int n_in,
                              void* d_out, int out_size, void* d_ws, size_t ws_size,
                              hipStream_t stream) {
  (void)in_sizes; (void)n_in; (void)out_size; (void)ws_size;
  const float* u = (const float*)d_in[0];
  const float* a = (const float*)d_in[1];
  const float* b = (const float*)d_in[2];
  const float* c = (const float*)d_in[3];
  float* f  = (float*)d_ws;                          // 256 f32 scale factors
  bf16* kbf = (bf16*)((char*)d_ws + 1024);           // 256 x KPADLEN bf16 (padded)

  norm_kernel<<<1, 256, 0, stream>>>(a, f);
  build_k<<<NKc, 64, 0, stream>>>(a, b, c, f, kbf);

  const size_t lds_bytes = (size_t)(16 * COLSTRIDE + KPADLEN) * sizeof(bf16);
  conv_wmma<<<2 * NKc, 256, lds_bytes, stream>>>(u, kbf, (float*)d_out);
}
